// GCN_7980049236110
// MI455X (gfx1250) — matrix operand order repaired and checked by reading, not compiled, not run on hardware
//
#include <hip/hip_runtime.h>
#include <hip/hip_bf16.h>
#include <math.h>

typedef __attribute__((ext_vector_type(16))) __bf16 v16bf;
typedef __attribute__((ext_vector_type(8)))  float  v8f;

#define GCN_N_NODES 100000
#define GCN_DIM_IN  500
#define GCN_DIM_H   128
#define GCN_DIM_OUT 40

__device__ __forceinline__ unsigned pack_bf16(float a, float b) {
#if __has_builtin(__builtin_amdgcn_cvt_pk_bf16_f32)
    typedef __attribute__((ext_vector_type(2))) __bf16 v2bf;
    union { v2bf v; unsigned u; } p;
    p.v = __builtin_amdgcn_cvt_pk_bf16_f32(a, b);   // v_cvt_pk_bf16_f32
    return p.u;
#else
    union { __bf16 h[2]; unsigned u; } p;
    p.h[0] = (__bf16)a;   // low half  = even K
    p.h[1] = (__bf16)b;   // high half = odd K
    return p.u;
#endif
}

// ---------------- degree / normalization ----------------

__global__ void init_deg_kernel(float* __restrict__ deg, int n) {
    int i = blockIdx.x * blockDim.x + threadIdx.x;
    if (i < n) deg[i] = 1.0f;  // self-loop
}

__global__ void count_deg_kernel(const int* __restrict__ col, float* __restrict__ deg, int nE) {
    int i = blockIdx.x * blockDim.x + threadIdx.x;
    int stride = gridDim.x * blockDim.x;
    for (; i < nE; i += stride) unsafeAtomicAdd(&deg[col[i]], 1.0f);
}

__global__ void deg_to_dinv_kernel(float* __restrict__ deg, int n) {
    int i = blockIdx.x * blockDim.x + threadIdx.x;
    if (i < n) deg[i] = rsqrtf(deg[i]);
}

// ---------------- bf16 WMMA GEMM with fragment-layout LDS ----------------
// C[MxN] = A[MxK] @ B[KxN], fp32 in/out, bf16 WMMA compute.
// Block = 256 threads = 8 waves; M-tile 128; NT = # of 16-wide N tiles.
// LDS tiles are PRE-SWIZZLED into the CDNA5 WMMA fragment layout so each lane
// loads its fragment as 32 contiguous bytes (2x ds_load_b128). All fragments
// are fetched before the WMMA chain so the matrix pipe issues back-to-back.

template <int NT>
__global__ __launch_bounds__(256)
void gemm_bf16_kernel(const float* __restrict__ A, const float* __restrict__ B,
                      float* __restrict__ C, int M, int N, int K) {
    constexpr int LN = 16 * NT;
    __shared__ __align__(32) unsigned As4[8][32][8];   // [wave][lane][dword]
    __shared__ __align__(32) unsigned Bs4[NT][32][8];  // [ntile][lane][dword]

    const int tid  = threadIdx.x;
    const int wave = tid >> 5;
    const int lane = tid & 31;
    const int row0 = blockIdx.x * 128;
    const int m    = lane & 15;
    const int kh   = lane >> 4;

    v8f acc[NT] = {};
    unsigned aw[8];      // staged A dwords (packed bf16 pairs)
    unsigned bw[NT];     // staged B dwords

    // ---- global fetch of one 32-wide k-slab into registers ----
    auto gload = [&](int kb) {
        #pragma unroll
        for (int j = 0; j < 8; ++j) {                 // A: 2048 dwords / 256 thr
            int l = tid + 256 * j;
            int kpair = l & 15, r = l >> 4;
            int gr = row0 + r, gc = kb + 2 * kpair;
            float f0 = 0.0f, f1 = 0.0f;
            if (gr < M && gc + 1 < K) {               // K is even for both layers
                const float2 t2 = *(const float2*)(A + (long)gr * K + gc);
                f0 = t2.x; f1 = t2.y;
            }
            aw[j] = pack_bf16(f0, f1);
        }
        #pragma unroll
        for (int j = 0; j < NT; ++j) {                // B: 16*LN dwords / 256 thr
            int d = tid + 256 * j;
            int c = d % LN;
            int kpair = d / LN;                       // 0..15
            int gk = kb + 2 * kpair;
            float f0 = 0.0f, f1 = 0.0f;
            if (c < N) {
                if (gk     < K) f0 = B[(long)gk * N + c];
                if (gk + 1 < K) f1 = B[(long)(gk + 1) * N + c];
            }
            bw[j] = pack_bf16(f0, f1);
        }
    };
    // ---- scatter staged registers into fragment-layout LDS ----
    auto lstore = [&]() {
        #pragma unroll
        for (int j = 0; j < 8; ++j) {
            int l = tid + 256 * j;
            int kpair = l & 15, r = l >> 4;
            int w = r >> 4, mm = r & 15;
            int lane_f = (((kpair >> 2) & 1) << 4) | mm;           // kh*16 + m
            int vva    = (((kpair >> 3) & 1) << 2) | (kpair & 3);  // 4g + p
            As4[w][lane_f][vva] = aw[j];
        }
        #pragma unroll
        for (int j = 0; j < NT; ++j) {
            int d = tid + 256 * j;
            int c = d % LN;
            int kpair = d / LN;
            int t = c >> 4;
            int lane_f = (((kpair >> 2) & 1) << 4) | (c & 15);
            Bs4[t][lane_f][(((kpair >> 3) & 1) << 2) | (kpair & 3)] = bw[j];
        }
    };

    gload(0);  // prologue
    for (int k0 = 0; k0 < K; k0 += 32) {
        lstore();
        __syncthreads();
        if (k0 + 32 < K) gload(k0 + 32);   // prefetch next slab behind WMMAs

        // fetch ALL fragments first (batched ds_load_b128), then WMMA chain
        v16bf a = *(const v16bf*)(&As4[wave][lane][0]);
        v16bf bfrag[NT];
        #pragma unroll
        for (int t = 0; t < NT; ++t)
            bfrag[t] = *(const v16bf*)(&Bs4[t][lane][0]);
        #pragma unroll
        for (int t = 0; t < NT; ++t)
            acc[t] = __builtin_amdgcn_wmma_f32_16x16x32_bf16(
                false, a, false, bfrag[t], (short)0, acc[t], false, false);
        __syncthreads();
    }

    // store C: VGPR v -> M = v + kh*8, N = lane&15 (ISA 7.12.2)
    #pragma unroll
    for (int t = 0; t < NT; ++t) {
        #pragma unroll
        for (int v = 0; v < 8; ++v) {
            int gm = row0 + wave * 16 + v + (kh << 3);
            int gn = t * 16 + m;
            if (gm < M && gn < N) C[(long)gm * N + gn] = acc[t][v];
        }
    }
}

// ---------------- edge scatter: out[c] += h[r] * dinv[r]*dinv[c] ----------------
// Work item = (edge, float4 chunk): global_load_b128 gather + 4 native fp32 atomics.

template <int F>
__global__ void scatter_kernel(const float* __restrict__ h, const float* __restrict__ dinv,
                               const int* __restrict__ row, const int* __restrict__ col,
                               float* __restrict__ out, int nE) {
    constexpr int C = F / 4;
    long gtid   = (long)blockIdx.x * blockDim.x + threadIdx.x;
    long stride = (long)gridDim.x * blockDim.x;
    long total  = (long)nE * C;
    for (long w = gtid; w < total; w += stride) {
        int e = (int)(w / C);
        int q = (int)(w % C);
        int r = row[e], c = col[e];
        float norm = dinv[r] * dinv[c];
        const float4 hv = *(const float4*)(h + (long)r * F + q * 4);
        float* o = out + (long)c * F + q * 4;
        unsafeAtomicAdd(o + 0, hv.x * norm);
        unsafeAtomicAdd(o + 1, hv.y * norm);
        unsafeAtomicAdd(o + 2, hv.z * norm);
        unsafeAtomicAdd(o + 3, hv.w * norm);
    }
}

// ---------------- fused self-loop + bias + ReLU (layer 1, float4) ----------------

__global__ void bias_relu_self_kernel(float* __restrict__ out1, const float* __restrict__ h,
                                      const float* __restrict__ dinv, const float* __restrict__ b,
                                      int total4) {  // N_NODES * 128 / 4
    int i = blockIdx.x * blockDim.x + threadIdx.x;
    if (i < total4) {
        int node = i >> 5;              // 32 float4 per node (F=128)
        int fq   = (i & 31) << 2;
        float di = dinv[node], sn = di * di;
        float4 o = ((const float4*)out1)[i];
        float4 hv = ((const float4*)h)[i];
        const float4 bv = *(const float4*)(b + fq);
        o.x = fmaxf(o.x + hv.x * sn + bv.x, 0.0f);
        o.y = fmaxf(o.y + hv.y * sn + bv.y, 0.0f);
        o.z = fmaxf(o.z + hv.z * sn + bv.z, 0.0f);
        o.w = fmaxf(o.w + hv.w * sn + bv.w, 0.0f);
        ((float4*)out1)[i] = o;
    }
}

// ---------------- self-loop + bias + log_softmax over 40 (wave per node) ----------------

__global__ void finalize_kernel(float* __restrict__ out, const float* __restrict__ h2,
                                const float* __restrict__ dinv, const float* __restrict__ b2,
                                int nNodes) {
    int gtid   = blockIdx.x * blockDim.x + threadIdx.x;
    int lane   = threadIdx.x & 31;
    int wid    = gtid >> 5;
    int nWaves = (gridDim.x * blockDim.x) >> 5;
    for (int node = wid; node < nNodes; node += nWaves) {
        long base = (long)node * 40;
        float di = dinv[node], sn = di * di;
        float v0 = out[base + lane] + h2[base + lane] * sn + b2[lane];
        float v1 = -INFINITY;
        if (lane < 8) v1 = out[base + 32 + lane] + h2[base + 32 + lane] * sn + b2[32 + lane];
        float mx = fmaxf(v0, v1);
        #pragma unroll
        for (int o = 16; o > 0; o >>= 1) mx = fmaxf(mx, __shfl_xor(mx, o, 32));
        float s = __expf(v0 - mx) + ((lane < 8) ? __expf(v1 - mx) : 0.0f);
        #pragma unroll
        for (int o = 16; o > 0; o >>= 1) s += __shfl_xor(s, o, 32);
        float lse = mx + __logf(s);
        out[base + lane] = v0 - lse;
        if (lane < 8) out[base + 32 + lane] = v1 - lse;
    }
}

// ---------------- driver ----------------

extern "C" void kernel_launch(void* const* d_in, const int* in_sizes, int n_in,
                              void* d_out, int out_size, void* d_ws, size_t ws_size,
                              hipStream_t stream) {
    const float* x  = (const float*)d_in[0];
    const int*   ei = (const int*)d_in[1];   // int32 (jax x64 disabled)
    const float* W1 = (const float*)d_in[2];
    const float* b1 = (const float*)d_in[3];
    const float* W2 = (const float*)d_in[4];
    const float* b2 = (const float*)d_in[5];
    float* out = (float*)d_out;

    const int nE   = in_sizes[1] / 2;
    const int* row = ei;
    const int* col = ei + nE;

    float* dinv = (float*)d_ws;
    float* h    = dinv + GCN_N_NODES;
    float* h1   = h + (size_t)GCN_N_NODES * GCN_DIM_H;
    float* h2   = h1 + (size_t)GCN_N_NODES * GCN_DIM_H;

    // symmetric normalization
    init_deg_kernel<<<(GCN_N_NODES + 255) / 256, 256, 0, stream>>>(dinv, GCN_N_NODES);
    count_deg_kernel<<<2048, 256, 0, stream>>>(col, dinv, nE);
    deg_to_dinv_kernel<<<(GCN_N_NODES + 255) / 256, 256, 0, stream>>>(dinv, GCN_N_NODES);

    // layer 1
    gemm_bf16_kernel<8><<<(GCN_N_NODES + 127) / 128, 256, 0, stream>>>(
        x, W1, h, GCN_N_NODES, GCN_DIM_H, GCN_DIM_IN);
    hipMemsetAsync(h1, 0, (size_t)GCN_N_NODES * GCN_DIM_H * sizeof(float), stream);
    scatter_kernel<GCN_DIM_H><<<8192, 256, 0, stream>>>(h, dinv, row, col, h1, nE);
    bias_relu_self_kernel<<<(GCN_N_NODES * GCN_DIM_H / 4 + 255) / 256, 256, 0, stream>>>(
        h1, h, dinv, b1, GCN_N_NODES * GCN_DIM_H / 4);

    // layer 2
    gemm_bf16_kernel<3><<<(GCN_N_NODES + 127) / 128, 256, 0, stream>>>(
        h1, W2, h2, GCN_N_NODES, GCN_DIM_OUT, GCN_DIM_H);
    hipMemsetAsync(out, 0, (size_t)GCN_N_NODES * GCN_DIM_OUT * sizeof(float), stream);
    scatter_kernel<GCN_DIM_OUT><<<8192, 256, 0, stream>>>(h2, dinv, row, col, out, nE);
    finalize_kernel<<<2048, 256, 0, stream>>>(out, h2, dinv, b2, GCN_N_NODES);
}
